// LSTMCell_678604833044
// MI455X (gfx1250) — matrix operand order repaired and checked by reading, not compiled, not run on hardware
//
#include <hip/hip_runtime.h>

// ---------------------------------------------------------------------------
// LSTM cell, fused:  z = [x|h] · [Wx|Wh]^T + (bx+bh),  gates, c_t/h_t.
// M=16384 rows, K=1536, N=4096 (4 gates x 1024).  bf16 WMMA, f32 accum.
// ---------------------------------------------------------------------------

typedef __bf16 bf16_t;
typedef __attribute__((ext_vector_type(16))) bf16_t v16bf;
typedef __attribute__((ext_vector_type(8)))  float  v8f;

#define M_TOT   16384
#define KK      1536
#define H_DIM   1024
#define KX      512

union Frag {
    uint4  u[2];   // 32 bytes = 16 bf16
    v16bf  v;
};

__device__ __forceinline__ unsigned short f2bf(float f) {
    unsigned u = __float_as_uint(f);
    unsigned r = u + 0x7FFFu + ((u >> 16) & 1u);   // round-to-nearest-even
    return (unsigned short)(r >> 16);
}

__device__ __forceinline__ unsigned pack2(float a, float b) {
    return (unsigned)f2bf(a) | ((unsigned)f2bf(b) << 16);
}

__device__ __forceinline__ float fsig(float x) {
    return 1.0f / (1.0f + __expf(-x));
}
__device__ __forceinline__ float ftanh(float x) {
    // tanh(x) = 1 - 2/(exp(2x)+1)  (exact limits at +-inf)
    return 1.0f - 2.0f / (__expf(2.0f * x) + 1.0f);
}

// --------------------------- prep: pack A = [x|h] ---------------------------
// Apack[m][k], k<512 from x, else from h.  8 elements per thread.
__global__ void pack_a_kernel(const float* __restrict__ x,
                              const float* __restrict__ h,
                              unsigned short* __restrict__ Apack) {
    size_t t    = (size_t)blockIdx.x * blockDim.x + threadIdx.x;
    size_t base = t * 8;                       // element index in Apack
    size_t m    = base / KK;
    int    k    = (int)(base % KK);
    const float* src = (k < KX) ? (x + m * (size_t)KX + k)
                                : (h + m * (size_t)H_DIM + (k - KX));
    float4 f0 = *(const float4*)(src);
    float4 f1 = *(const float4*)(src + 4);
    uint4 o;
    o.x = pack2(f0.x, f0.y);
    o.y = pack2(f0.z, f0.w);
    o.z = pack2(f1.x, f1.y);
    o.w = pack2(f1.z, f1.w);
    *(uint4*)(Apack + base) = o;
}

// --------------------------- prep: pack B = [Wx|Wh] -------------------------
// Bpack[g][k], g = gate*1024 + row; k<512 from Wx_gate, else Wh_gate.
__global__ void pack_b_kernel(const float* __restrict__ wii, const float* __restrict__ wif,
                              const float* __restrict__ wig, const float* __restrict__ wio,
                              const float* __restrict__ whi, const float* __restrict__ whf,
                              const float* __restrict__ whg, const float* __restrict__ who,
                              unsigned short* __restrict__ Bpack) {
    size_t t    = (size_t)blockIdx.x * blockDim.x + threadIdx.x;
    size_t base = t * 8;
    size_t g    = base / KK;
    int    k    = (int)(base % KK);
    int gate = (int)(g >> 10);
    int gl   = (int)(g & 1023);
    const float* Wx[4] = {wii, wif, wig, wio};
    const float* Wh[4] = {whi, whf, whg, who};
    const float* src = (k < KX) ? (Wx[gate] + (size_t)gl * KX + k)
                                : (Wh[gate] + (size_t)gl * H_DIM + (k - KX));
    float4 f0 = *(const float4*)(src);
    float4 f1 = *(const float4*)(src + 4);
    uint4 o;
    o.x = pack2(f0.x, f0.y);
    o.y = pack2(f0.z, f0.w);
    o.z = pack2(f1.x, f1.y);
    o.w = pack2(f1.z, f1.w);
    *(uint4*)(Bpack + base) = o;
}

// --------------------------- prep: fold biases ------------------------------
__global__ void pack_bias_kernel(const float* __restrict__ bii, const float* __restrict__ bif,
                                 const float* __restrict__ big, const float* __restrict__ bio,
                                 const float* __restrict__ bhi, const float* __restrict__ bhf,
                                 const float* __restrict__ bhg, const float* __restrict__ bho,
                                 float* __restrict__ bias) {
    int g    = blockIdx.x * blockDim.x + threadIdx.x;   // 0..4095
    int gate = g >> 10;
    int gl   = g & 1023;
    const float* bx[4] = {bii, bif, big, bio};
    const float* bh[4] = {bhi, bhf, bhg, bho};
    bias[g] = bx[gate][gl] + bh[gate][gl];
}

// --------------------------- main: GEMM + gates -----------------------------
// Block: 256 threads = 8 waves.  Wave tile: 32 rows x 32 cols x 4 gates
// (16 accumulators v8f).  Block tile: 256 rows x 32 cols (per gate).
// grid = (1024/32, 16384/256) = (32, 64).
__global__ void __launch_bounds__(256)
lstm_gemm_kernel(const unsigned short* __restrict__ Apack,
                 const unsigned short* __restrict__ Bpack,
                 const float* __restrict__ bias,
                 const float* __restrict__ c_in,
                 float* __restrict__ c_out,
                 float* __restrict__ h_out) {
    const int lane = threadIdx.x & 31;
    const int wave = threadIdx.x >> 5;
    const int hi   = lane >> 4;      // half-wave select (K-split per ISA layout)
    const int lrow = lane & 15;      // M (A frag) / N (B frag) index within tile

    const int m0 = blockIdx.y * 256 + wave * 32;   // wave's first row
    const int j0 = blockIdx.x * 32;                // per-gate column base

    v8f acc[2][2][4] = {};   // [mi][nj][gate]

    for (int k0 = 0; k0 < KK; k0 += 32) {
        // A fragments: lane<16 holds K = k0+0..7 & k0+16..23 of row m;
        //              lane>=16 holds K = k0+8..15 & k0+24..31 of row m.
        Frag a[2];
        const int ak = k0 + hi * 8;
#pragma unroll
        for (int mi = 0; mi < 2; ++mi) {
            const unsigned short* p =
                Apack + (size_t)(m0 + mi * 16 + lrow) * KK + ak;
            a[mi].u[0] = *(const uint4*)(p);
            a[mi].u[1] = *(const uint4*)(p + 16);
        }
        // B fragments: lane<16 holds K = k0+0..7 & k0+16..23 of column n;
        //              lane>=16 holds K = k0+8..15 & k0+24..31 of column n.
        Frag b[2][4];
        const int bk = k0 + hi * 8;
#pragma unroll
        for (int nj = 0; nj < 2; ++nj) {
#pragma unroll
            for (int g = 0; g < 4; ++g) {
                const unsigned short* p =
                    Bpack + (size_t)(g * H_DIM + j0 + nj * 16 + lrow) * KK + bk;
                b[nj][g].u[0] = *(const uint4*)(p);
                b[nj][g].u[1] = *(const uint4*)(p + 16);
            }
        }
#pragma unroll
        for (int mi = 0; mi < 2; ++mi)
#pragma unroll
            for (int nj = 0; nj < 2; ++nj)
#pragma unroll
                for (int g = 0; g < 4; ++g)
                    acc[mi][nj][g] = __builtin_amdgcn_wmma_f32_16x16x32_bf16(
                        /*neg_a=*/false, a[mi].v,
                        /*neg_b=*/false, b[nj][g].v,
                        /*c_mod=*/(short)0, acc[mi][nj][g],
                        /*reuse_a=*/false, /*reuse_b=*/false);
    }

    // Epilogue — fully in-register: for fixed (lane, slot v) all four gates'
    // accumulators hold the SAME (m, j) element (C layout: M = v + 8*hi,
    // N = lane&15), so no LDS exchange is needed.
#pragma unroll
    for (int nj = 0; nj < 2; ++nj) {
        const int jg = j0 + nj * 16 + lrow;
        const float bi = bias[jg];
        const float bf = bias[H_DIM + jg];
        const float bg = bias[2 * H_DIM + jg];
        const float bo = bias[3 * H_DIM + jg];
#pragma unroll
        for (int mi = 0; mi < 2; ++mi) {
#pragma unroll
            for (int v = 0; v < 8; ++v) {
                const int m = m0 + mi * 16 + hi * 8 + v;
                const float zi = acc[mi][nj][0][v] + bi;
                const float zf = acc[mi][nj][1][v] + bf;
                const float zg = acc[mi][nj][2][v] + bg;
                const float zo = acc[mi][nj][3][v] + bo;
                const float it = fsig(zi);
                const float ft = fsig(zf);
                const float gt = ftanh(zg);
                const float ot = fsig(zo);
                const size_t off = (size_t)m * H_DIM + jg;
                const float ct = ft * c_in[off] + it * gt;
                c_out[off] = ct;
                h_out[off] = ot * ftanh(ct);
            }
        }
    }
}

// ---------------------------------------------------------------------------
extern "C" void kernel_launch(void* const* d_in, const int* in_sizes, int n_in,
                              void* d_out, int out_size, void* d_ws, size_t ws_size,
                              hipStream_t stream) {
    const float* x = (const float*)d_in[0];
    const float* c = (const float*)d_in[1];
    const float* h = (const float*)d_in[2];
    // setup_inputs order: W_ii,b_ii, W_if_,b_if_, W_ig,b_ig, W_io,b_io,
    //                     W_hi,b_hi, W_hf,b_hf, W_hg,b_hg, W_ho,b_ho
    const float* W_ii = (const float*)d_in[3];
    const float* b_ii = (const float*)d_in[4];
    const float* W_if = (const float*)d_in[5];
    const float* b_if = (const float*)d_in[6];
    const float* W_ig = (const float*)d_in[7];
    const float* b_ig = (const float*)d_in[8];
    const float* W_io = (const float*)d_in[9];
    const float* b_io = (const float*)d_in[10];
    const float* W_hi = (const float*)d_in[11];
    const float* b_hi = (const float*)d_in[12];
    const float* W_hf = (const float*)d_in[13];
    const float* b_hf = (const float*)d_in[14];
    const float* W_hg = (const float*)d_in[15];
    const float* b_hg = (const float*)d_in[16];
    const float* W_ho = (const float*)d_in[17];
    const float* b_ho = (const float*)d_in[18];

    // Workspace layout: Apack 48MB | Bpack 12MB | bias 16KB
    char* ws = (char*)d_ws;
    unsigned short* Apack = (unsigned short*)(ws);
    unsigned short* Bpack = (unsigned short*)(ws + (size_t)M_TOT * KK * 2);
    float*          biasv = (float*)(ws + (size_t)M_TOT * KK * 2 + (size_t)4096 * KK * 2);

    // prep
    pack_a_kernel<<<(M_TOT * (size_t)KK / 8) / 256, 256, 0, stream>>>(x, h, Apack);
    pack_b_kernel<<<(4096 * (size_t)KK / 8) / 256, 256, 0, stream>>>(
        W_ii, W_if, W_ig, W_io, W_hi, W_hf, W_hg, W_ho, Bpack);
    pack_bias_kernel<<<4096 / 256, 256, 0, stream>>>(
        b_ii, b_if, b_ig, b_io, b_hi, b_hf, b_hg, b_ho, biasv);

    // GEMM + gates
    float* c_out = (float*)d_out;
    float* h_out = c_out + (size_t)M_TOT * H_DIM;
    dim3 grid(H_DIM / 32, M_TOT / 256);
    lstm_gemm_kernel<<<grid, 256, 0, stream>>>(Apack, Bpack, biasv, c, c_out, h_out);
}
